// AttentionStoreActivationPrune_58110907515136
// MI455X (gfx1250) — matrix operand order repaired and checked by reading, not compiled, not run on hardware
//
#include <hip/hip_runtime.h>
#include <hip/hip_bf16.h>
#include <math.h>

// ---------------------------------------------------------------------------
// MHA forward for B=4, S=1024, HID=768, NH=12, D=64 on gfx1250 (wave32, WMMA)
// Matmuls via v_wmma_f32_16x16x32_bf16 (bf16 operands, f32 accumulate).
// Attention is fused flash-style (probs written exactly once). Q/K/V tiles are
// staged into LDS by the Tensor Data Mover with hardware row padding and
// double-buffered TENSORcnt pipelining; the 201 MB probs output is written by
// TENSOR_STORE_FROM_LDS riding the same in-order tensor counter.
// Outputs: out [4,1024,768] f32, then probs [4,12,1024,1024] f32 (concat).
// ---------------------------------------------------------------------------

typedef __attribute__((ext_vector_type(16))) __bf16 v16bf;
typedef __attribute__((ext_vector_type(8)))  float  v8f;
typedef __attribute__((ext_vector_type(4)))  unsigned int u32x4;
typedef __attribute__((ext_vector_type(8)))  int    i32x8;
typedef __attribute__((ext_vector_type(4)))  int    i32x4;

#define BDIM 4
#define SDIM 1024
#define HID 768
#define NH 12
#define DH 64
#define ROWS (BDIM * SDIM)   // 4096

#if __has_builtin(__builtin_amdgcn_tensor_load_to_lds)
#define HAVE_TDM 1
#if __has_include(<hip/amd_detail/amd_gfx1250_TDM.h>)
#define TDM_ARITY 6   // therock-10.0 headers -> 6-arg builtin
#else
#define TDM_ARITY 5   // ROCm 7.2 clang-22 -> 5-arg builtin
#endif
#else
#define HAVE_TDM 0
#endif

#if HAVE_TDM && __has_builtin(__builtin_amdgcn_tensor_store_from_lds)
#define HAVE_TDM_STORE 1
#else
#define HAVE_TDM_STORE 0
#endif

// ---------------- WMMA fragment loaders (wave32; contiguous k = 16*(L>>4) + e in A and B) ------

// A matrix 16x32 bf16, row-major source in LDS with leading dim `ld`.
// lane L: M = L&15, group g = L>>4 ; element e -> K = g*16 + e
__device__ __forceinline__ v16bf load_a_frag(const __bf16* lds, int row0, int k0, int ld) {
    const int lane = threadIdx.x & 31;
    const int m = lane & 15;
    const int g = lane >> 4;
    const __bf16* p = lds + (row0 + m) * ld + k0;
    v16bf a;
#pragma unroll
    for (int e = 0; e < 16; ++e) {
        const int kk = g * 16 + e;
        a[e] = p[kk];
    }
    return a;
}

// B matrix 32x16 bf16 where LDS tile is stored [N][K] row-major.
// lane L: N = L&15, g = L>>4 ; element e -> K = g*16 + e
__device__ __forceinline__ v16bf load_b_frag_nk(const __bf16* lds, int n0, int k0, int ld) {
    const int lane = threadIdx.x & 31;
    const int n = lane & 15;
    const int g = lane >> 4;
    const __bf16* p = lds + (n0 + n) * ld + k0 + g * 16;
    v16bf b;
#pragma unroll
    for (int e = 0; e < 16; ++e) b[e] = p[e];
    return b;
}

// B matrix 32x16 bf16 where LDS tile is stored [K][N] row-major (V tiles).
__device__ __forceinline__ v16bf load_b_frag_kn(const __bf16* lds, int k0, int n0, int ld) {
    const int lane = threadIdx.x & 31;
    const int n = lane & 15;
    const int g = lane >> 4;
    const __bf16* p = lds + (k0 + g * 16) * ld + n0 + n;
    v16bf b;
#pragma unroll
    for (int e = 0; e < 16; ++e) b[e] = p[e * ld];
    return b;
}

__device__ __forceinline__ v8f wmma_bf16(v16bf a, v16bf b, v8f c) {
    return __builtin_amdgcn_wmma_f32_16x16x32_bf16(false, a, false, b, (short)0, c,
                                                   false, false);
}

// ---------------- Tensor Data Mover helpers ---------------------------------
__device__ __forceinline__ unsigned lds_off(const void* p) {
    // generic pointers to LDS carry the workgroup-relative offset in addr[31:0]
    return (unsigned)(unsigned long long)(uintptr_t)p;
}

#if HAVE_TDM
// 64x64 bf16 tile (global row stride HID elems) -> LDS with 16B row padding
// (stride 72 elements, matches the WMMA staging layout).
__device__ __forceinline__ void tdm_load_tile64(const __bf16* gsrc, unsigned ldsoff) {
    const unsigned long long ga = (unsigned long long)(uintptr_t)gsrc;
    u32x4 g0;
    g0[0] = 1u;                                     // count=1 (one valid D#)
    g0[1] = ldsoff;                                 // lds_addr (bytes)
    g0[2] = (unsigned)ga;                           // global_addr[31:0]
    g0[3] = (unsigned)((ga >> 32) & 0x01FFFFFFu)    // global_addr[56:32]
            | (2u << 30);                           // type=2 ("image")
    i32x8 g1;
    g1[0] = (int)((1u << 16)     // data_size=1 -> 2-byte elements
                | (1u << 20)     // pad_enable
                | (4u << 22)     // pad_interval: 32 DWORDs (=128B = one row)
                | (3u << 25));   // pad_amount: 4 DWORDs (=16B -> stride 72)
    g1[1] = (int)(64u << 16);    // tensor_dim0 = 64
    g1[2] = (int)(64u << 16);    // tensor_dim1 = 64
    g1[3] = (int)(64u << 16);    // tile_dim0 = 64
    g1[4] = 64;                  // tile_dim1 = 64 ; tile_dim2 = 0
    g1[5] = HID;                 // tensor_dim0_stride = 768 elements
    g1[6] = 0;
    g1[7] = 0;
    const i32x4 z = {0, 0, 0, 0};
#if TDM_ARITY == 6
    const i32x8 z8 = {0, 0, 0, 0, 0, 0, 0, 0};
    __builtin_amdgcn_tensor_load_to_lds(g0, g1, z, z, z8, 0);
#else
    __builtin_amdgcn_tensor_load_to_lds(g0, g1, z, z, 0);
#endif
}
#endif

#if HAVE_TDM_STORE
// contiguous 64x64 f32 LDS tile -> global tile with row stride SDIM elements
__device__ __forceinline__ void tdm_store_tile64_f32(float* gdst, unsigned ldsoff) {
    const unsigned long long ga = (unsigned long long)(uintptr_t)gdst;
    u32x4 g0;
    g0[0] = 1u;
    g0[1] = ldsoff;
    g0[2] = (unsigned)ga;
    g0[3] = (unsigned)((ga >> 32) & 0x01FFFFFFu) | (2u << 30);
    i32x8 g1;
    g1[0] = (int)(2u << 16);     // data_size=2 -> 4-byte elements, no pad
    g1[1] = (int)(64u << 16);    // tensor_dim0 = 64
    g1[2] = (int)(64u << 16);    // tensor_dim1 = 64
    g1[3] = (int)(64u << 16);    // tile_dim0 = 64
    g1[4] = 64;                  // tile_dim1 = 64
    g1[5] = SDIM;                // tensor_dim0_stride = 1024 elements
    g1[6] = 0;
    g1[7] = 0;
    const i32x4 z = {0, 0, 0, 0};
#if TDM_ARITY == 6
    const i32x8 z8 = {0, 0, 0, 0, 0, 0, 0, 0};
    __builtin_amdgcn_tensor_store_from_lds(g0, g1, z, z, z8, 0);
#else
    __builtin_amdgcn_tensor_store_from_lds(g0, g1, z, z, 0);
#endif
}
#endif

// ---------------- GEMM: Out[m,n] = sum_k A[m,k]*W[n,k] + bias[n] -------------
template <bool A_BF16, bool OUT_F32>
__global__ void __launch_bounds__(128)
gemm_nt_bias(const void* __restrict__ Aptr, const float* __restrict__ W,
             const float* __restrict__ bias, void* __restrict__ Out,
             int M, int N, int K) {
    constexpr int LDT = 40;
    __shared__ __bf16 As[64 * LDT];
    __shared__ __bf16 Bs[64 * LDT];

    const int tid  = threadIdx.x;
    const int lane = tid & 31;
    const int wave = tid >> 5;
    const int wm = (wave >> 1) * 32;
    const int wn = (wave & 1) * 32;
    const int m0 = blockIdx.y * 64;
    const int n0 = blockIdx.x * 64;

    v8f acc[2][2] = {{{0.f}, {0.f}}, {{0.f}, {0.f}}};

    for (int k0 = 0; k0 < K; k0 += 32) {
#pragma unroll
        for (int i = 0; i < 16; ++i) {
            const int idx = tid + i * 128;
            const int r = idx >> 5, c = idx & 31;
            float av;
            if (A_BF16)
                av = (float)((const __bf16*)Aptr)[(size_t)(m0 + r) * K + k0 + c];
            else
                av = ((const float*)Aptr)[(size_t)(m0 + r) * K + k0 + c];
            As[r * LDT + c] = (__bf16)av;
            Bs[r * LDT + c] = (__bf16)W[(size_t)(n0 + r) * K + k0 + c];
        }
        __syncthreads();

        // prefetch next W tile rows (global_prefetch_b8)
        if (k0 + 32 < K && tid < 64)
            __builtin_prefetch(&W[(size_t)(n0 + tid) * K + k0 + 32], 0, 1);

        const v16bf a0 = load_a_frag(As, wm + 0, 0, LDT);
        const v16bf a1 = load_a_frag(As, wm + 16, 0, LDT);
        const v16bf b0 = load_b_frag_nk(Bs, wn + 0, 0, LDT);
        const v16bf b1 = load_b_frag_nk(Bs, wn + 16, 0, LDT);
        acc[0][0] = wmma_bf16(a0, b0, acc[0][0]);
        acc[0][1] = wmma_bf16(a0, b1, acc[0][1]);
        acc[1][0] = wmma_bf16(a1, b0, acc[1][0]);
        acc[1][1] = wmma_bf16(a1, b1, acc[1][1]);
        __syncthreads();
    }

    const int n = lane & 15;
    const int g = lane >> 4;
#pragma unroll
    for (int ti = 0; ti < 2; ++ti)
#pragma unroll
        for (int tj = 0; tj < 2; ++tj) {
            const int colg = n0 + wn + tj * 16 + n;
            const float bv = bias[colg];
#pragma unroll
            for (int v = 0; v < 8; ++v) {
                const int rowg = m0 + wm + ti * 16 + g * 8 + v;
                const float val = acc[ti][tj][v] + bv;
                if (OUT_F32)
                    ((float*)Out)[(size_t)rowg * N + colg] = val;
                else
                    ((__bf16*)Out)[(size_t)rowg * N + colg] = (__bf16)val;
            }
        }
}

// ---------------- fused attention ------------------------------------------
__global__ void __launch_bounds__(128)
fused_attn(const __bf16* __restrict__ qb, const __bf16* __restrict__ kb,
           const __bf16* __restrict__ vb, float* __restrict__ probs,
           __bf16* __restrict__ ctxb) {
    constexpr int LDT = 72;  // 64 + 8 pad (matches TDM pad descriptor)
    __shared__ __bf16 Qs[64 * LDT];
    __shared__ __bf16 Ks[2][64 * LDT];
    __shared__ __bf16 Vs[2][64 * LDT];
    __shared__ __bf16 Ps[64 * LDT];
#if HAVE_TDM_STORE
    __shared__ float  Pf[2][64 * 64];   // contiguous f32 tile for TDM store
#endif
    __shared__ float  red_m[64][33];
    __shared__ float  red_s[64][33];
    __shared__ float  row_m[64];
    __shared__ float  row_inv[64];

    const int tid  = threadIdx.x;
    const int lane = tid & 31;
    const int wave = tid >> 5;
    const int wm = (wave >> 1) * 32;
    const int wn = (wave & 1) * 32;
    const int nl = lane & 15;
    const int g  = lane >> 4;
    const int m0 = blockIdx.x * 64;
    const int bh = blockIdx.y;
    const int b  = bh / NH;
    const int h  = bh % NH;
    const float scale = 0.125f;  // 1/sqrt(64)
    constexpr int NT = SDIM / 64;

    const __bf16* qbase = qb + (size_t)(b * SDIM) * HID + h * DH;
    const __bf16* kbase = kb + (size_t)(b * SDIM) * HID + h * DH;
    const __bf16* vbase = vb + (size_t)(b * SDIM) * HID + h * DH;

#if HAVE_TDM
    // TDM-stage Q slab once; retired together with K(0) by the first wait
    if (wave == 0) tdm_load_tile64(qbase + (size_t)m0 * HID, lds_off(&Qs[0]));
#else
#pragma unroll
    for (int i = 0; i < 32; ++i) {
        const int idx = tid + i * 128;
        const int r = idx >> 6, c = idx & 63;
        Qs[r * LDT + c] = qbase[(size_t)(m0 + r) * HID + c];
    }
    auto stage_tile = [&](const __bf16* src, __bf16* dst) {
#pragma unroll
        for (int i = 0; i < 32; ++i) {
            const int idx = tid + i * 128;
            const int r = idx >> 6, c = idx & 63;
            dst[r * LDT + c] = src[(size_t)r * HID + c];
        }
    };
#endif

    // ---------------- pass 1: online row max / sum --------------------------
    float m_loc[16], s_loc[16];
#pragma unroll
    for (int i = 0; i < 16; ++i) { m_loc[i] = -INFINITY; s_loc[i] = 0.f; }

#if HAVE_TDM
    if (wave == 0) tdm_load_tile64(kbase, lds_off(&Ks[0][0]));
#endif
    for (int jt = 0; jt < NT; ++jt) {
        const int j0 = jt * 64;
#if HAVE_TDM
        if (wave == 0) {
            if (jt + 1 < NT) {
                tdm_load_tile64(kbase + (size_t)(j0 + 64) * HID,
                                lds_off(&Ks[(jt + 1) & 1][0]));
                __builtin_amdgcn_s_wait_tensorcnt(1);  // (Q and) K(jt) complete
            } else {
                __builtin_amdgcn_s_wait_tensorcnt(0);
            }
        }
        __syncthreads();
        const __bf16* Kcur = Ks[jt & 1];
#else
        stage_tile(kbase + (size_t)j0 * HID, Ks[0]);
        __syncthreads();
        const __bf16* Kcur = Ks[0];
#endif

        v8f acc[2][2] = {{{0.f}, {0.f}}, {{0.f}, {0.f}}};
#pragma unroll
        for (int k0 = 0; k0 < DH; k0 += 32) {
            const v16bf a0 = load_a_frag(Qs, wm + 0, k0, LDT);
            const v16bf a1 = load_a_frag(Qs, wm + 16, k0, LDT);
            const v16bf b0 = load_b_frag_nk(Kcur, wn + 0, k0, LDT);
            const v16bf b1 = load_b_frag_nk(Kcur, wn + 16, k0, LDT);
            acc[0][0] = wmma_bf16(a0, b0, acc[0][0]);
            acc[0][1] = wmma_bf16(a0, b1, acc[0][1]);
            acc[1][0] = wmma_bf16(a1, b0, acc[1][0]);
            acc[1][1] = wmma_bf16(a1, b1, acc[1][1]);
        }

#pragma unroll
        for (int ti = 0; ti < 2; ++ti)
#pragma unroll
            for (int v = 0; v < 8; ++v) {
                const int i = ti * 8 + v;
                const float x0 = acc[ti][0][v] * scale;
                const float x1 = acc[ti][1][v] * scale;
                const float mnew = fmaxf(m_loc[i], fmaxf(x0, x1));
                s_loc[i] = s_loc[i] * __expf(m_loc[i] - mnew) +
                           __expf(x0 - mnew) + __expf(x1 - mnew);
                m_loc[i] = mnew;
            }
        __syncthreads();  // readers done before this buffer is re-filled
    }

    // combine partials across the 32 (wave-col, n-lane) slots per row
    const int slot = (wave & 1) * 16 + nl;
#pragma unroll
    for (int ti = 0; ti < 2; ++ti)
#pragma unroll
        for (int v = 0; v < 8; ++v) {
            const int r = wm + ti * 16 + g * 8 + v;
            red_m[r][slot] = m_loc[ti * 8 + v];
            red_s[r][slot] = s_loc[ti * 8 + v];
        }
    __syncthreads();
    if (tid < 64) {
        float m = -INFINITY;
#pragma unroll
        for (int s = 0; s < 32; ++s) m = fmaxf(m, red_m[tid][s]);
        float sum = 0.f;
#pragma unroll
        for (int s = 0; s < 32; ++s) sum += red_s[tid][s] * __expf(red_m[tid][s] - m);
        row_m[tid] = m;
        row_inv[tid] = 1.0f / sum;
    }
    __syncthreads();

    float mr[16], ir[16];
#pragma unroll
    for (int ti = 0; ti < 2; ++ti)
#pragma unroll
        for (int v = 0; v < 8; ++v) {
            const int r = wm + ti * 16 + g * 8 + v;
            mr[ti * 8 + v] = row_m[r];
            ir[ti * 8 + v] = row_inv[r];
        }

    // ---------------- pass 2: probs out (TDM store) + ctx = P.V -------------
    v8f ctx[2][2] = {{{0.f}, {0.f}}, {{0.f}, {0.f}}};

#if HAVE_TDM
    if (wave == 0) {
        tdm_load_tile64(kbase, lds_off(&Ks[0][0]));
        tdm_load_tile64(vbase, lds_off(&Vs[0][0]));
    }
#endif
    for (int jt = 0; jt < NT; ++jt) {
        const int j0 = jt * 64;
#if HAVE_TDM
        if (wave == 0) {
            if (jt + 1 < NT) {
                tdm_load_tile64(kbase + (size_t)(j0 + 64) * HID,
                                lds_off(&Ks[(jt + 1) & 1][0]));
                tdm_load_tile64(vbase + (size_t)(j0 + 64) * HID,
                                lds_off(&Vs[(jt + 1) & 1][0]));
                // in-order retire: K(jt), V(jt) and (if issued) store(jt-1)
                __builtin_amdgcn_s_wait_tensorcnt(2);
            } else {
                __builtin_amdgcn_s_wait_tensorcnt(0);
            }
        }
        __syncthreads();
        const __bf16* Kcur = Ks[jt & 1];
        const __bf16* Vcur = Vs[jt & 1];
#else
        stage_tile(kbase + (size_t)j0 * HID, Ks[0]);
        stage_tile(vbase + (size_t)j0 * HID, Vs[0]);
        __syncthreads();
        const __bf16* Kcur = Ks[0];
        const __bf16* Vcur = Vs[0];
#endif

        v8f acc[2][2] = {{{0.f}, {0.f}}, {{0.f}, {0.f}}};
#pragma unroll
        for (int k0 = 0; k0 < DH; k0 += 32) {
            const v16bf a0 = load_a_frag(Qs, wm + 0, k0, LDT);
            const v16bf a1 = load_a_frag(Qs, wm + 16, k0, LDT);
            const v16bf b0 = load_b_frag_nk(Kcur, wn + 0, k0, LDT);
            const v16bf b1 = load_b_frag_nk(Kcur, wn + 16, k0, LDT);
            acc[0][0] = wmma_bf16(a0, b0, acc[0][0]);
            acc[0][1] = wmma_bf16(a0, b1, acc[0][1]);
            acc[1][0] = wmma_bf16(a1, b0, acc[1][0]);
            acc[1][1] = wmma_bf16(a1, b1, acc[1][1]);
        }

        // normalize; stage P as bf16 (WMMA A operand) and f32 (TDM store tile)
#pragma unroll
        for (int ti = 0; ti < 2; ++ti)
#pragma unroll
            for (int tj = 0; tj < 2; ++tj) {
                const int col = wn + tj * 16 + nl;
#pragma unroll
                for (int v = 0; v < 8; ++v) {
                    const int row = wm + ti * 16 + g * 8 + v;
                    const float p = __expf(acc[ti][tj][v] * scale - mr[ti * 8 + v]) *
                                    ir[ti * 8 + v];
                    Ps[row * LDT + col] = (__bf16)p;
#if HAVE_TDM_STORE
                    Pf[jt & 1][row * 64 + col] = p;
#else
                    probs[((size_t)bh * SDIM + m0 + row) * SDIM + j0 + col] = p;
#endif
                }
            }
        __syncthreads();

#if HAVE_TDM_STORE
        // one DMA store for the finished 64x64 probs tile
        if (wave == 0)
            tdm_store_tile64_f32(probs + ((size_t)bh * SDIM + m0) * SDIM + j0,
                                 lds_off(&Pf[jt & 1][0]));
#endif

        // ctx += P(64x64) * V(64x64)
#pragma unroll
        for (int k0 = 0; k0 < 64; k0 += 32) {
            const v16bf a0 = load_a_frag(Ps, wm + 0, k0, LDT);
            const v16bf a1 = load_a_frag(Ps, wm + 16, k0, LDT);
            const v16bf b0 = load_b_frag_kn(Vcur, k0, wn + 0, LDT);
            const v16bf b1 = load_b_frag_kn(Vcur, k0, wn + 16, LDT);
            ctx[0][0] = wmma_bf16(a0, b0, ctx[0][0]);
            ctx[0][1] = wmma_bf16(a0, b1, ctx[0][1]);
            ctx[1][0] = wmma_bf16(a1, b0, ctx[1][0]);
            ctx[1][1] = wmma_bf16(a1, b1, ctx[1][1]);
        }
        __syncthreads();
    }
    // final TDM store drains via S_ENDPGM's implicit wait-idle

    // store ctx slab (bf16) into workspace, merged-head layout [B,S,HID]
#pragma unroll
    for (int ti = 0; ti < 2; ++ti)
#pragma unroll
        for (int tj = 0; tj < 2; ++tj) {
            const int colg = wn + tj * 16 + nl;
#pragma unroll
            for (int v = 0; v < 8; ++v) {
                const int rowg = m0 + wm + ti * 16 + g * 8 + v;
                ctxb[(size_t)(b * SDIM + rowg) * HID + h * DH + colg] =
                    (__bf16)ctx[ti][tj][v];
            }
        }
}

// ---------------------------------------------------------------------------
extern "C" void kernel_launch(void* const* d_in, const int* in_sizes, int n_in,
                              void* d_out, int out_size, void* d_ws, size_t ws_size,
                              hipStream_t stream) {
    const float* hs = (const float*)d_in[0];
    const float* Wq = (const float*)d_in[1];
    const float* bq = (const float*)d_in[2];
    const float* Wk = (const float*)d_in[3];
    const float* bk = (const float*)d_in[4];
    const float* Wv = (const float*)d_in[5];
    const float* bv = (const float*)d_in[6];
    const float* Wo = (const float*)d_in[7];
    const float* bo = (const float*)d_in[8];

    float* out   = (float*)d_out;                       // [4096, 768]
    float* probs = (float*)d_out + (size_t)ROWS * HID;  // [48, 1024, 1024]

    __bf16* qb   = (__bf16*)d_ws;
    __bf16* kb   = qb + (size_t)ROWS * HID;
    __bf16* vb   = kb + (size_t)ROWS * HID;
    __bf16* ctxb = vb + (size_t)ROWS * HID;

    const dim3 blk(128);

    gemm_nt_bias<false, false><<<dim3(HID / 64, ROWS / 64), blk, 0, stream>>>(
        hs, Wq, bq, qb, ROWS, HID, HID);
    gemm_nt_bias<false, false><<<dim3(HID / 64, ROWS / 64), blk, 0, stream>>>(
        hs, Wk, bk, kb, ROWS, HID, HID);
    gemm_nt_bias<false, false><<<dim3(HID / 64, ROWS / 64), blk, 0, stream>>>(
        hs, Wv, bv, vb, ROWS, HID, HID);

    fused_attn<<<dim3(SDIM / 64, BDIM * NH), blk, 0, stream>>>(qb, kb, vb, probs,
                                                               ctxb);

    gemm_nt_bias<true, true><<<dim3(HID / 64, ROWS / 64), blk, 0, stream>>>(
        ctxb, Wo, bo, out, ROWS, HID, HID);
}
